// GumbelCusteringBMUOnly_1460288881072
// MI455X (gfx1250) — matrix operand order repaired and checked by reading, not compiled, not run on hardware
//
#include <hip/hip_runtime.h>
#include <hip/hip_bf16.h>
#include <stdint.h>

// Problem constants (match reference)
#define B_  16384
#define D_  1024
#define K_  8192
#define H_  90
#define HP_ 96     // H padded up to a multiple of 32 for WMMA K-dim

typedef __attribute__((ext_vector_type(16))) __bf16 v16bf;
typedef __attribute__((ext_vector_type(8)))  __bf16 v8bf;
typedef __attribute__((ext_vector_type(8)))  float  v8f;
typedef __attribute__((ext_vector_type(4)))  unsigned int v4u;
typedef __attribute__((ext_vector_type(8)))  int    v8i;
typedef __attribute__((ext_vector_type(4)))  int    v4i;

// ---------------------------------------------------------------------------
// helpers
// ---------------------------------------------------------------------------
__device__ __forceinline__ __bf16 to_bf16(float f) {
    unsigned u = __float_as_uint(f);
    unsigned r = (u + 0x7FFFu + ((u >> 16) & 1u)) >> 16;   // RNE
    unsigned short s = (unsigned short)r;
    return *(__bf16*)&s;
}

// monotonic map float -> u32 (order preserving over all floats)
__device__ __forceinline__ unsigned mono_f32(float f) {
    unsigned u = __float_as_uint(f);
    return (u & 0x80000000u) ? ~u : (u | 0x80000000u);
}
__device__ __forceinline__ unsigned long long pack_min_key(float d, unsigned col) {
    return ((unsigned long long)mono_f32(d) << 32) | (unsigned long long)col;           // min -> smallest col on tie
}
__device__ __forceinline__ unsigned long long pack_max_key(float d, unsigned col) {
    return ((unsigned long long)mono_f32(d) << 32) | (unsigned long long)(0xFFFFFFFFu - col); // max -> smallest col on tie
}

__device__ __forceinline__ unsigned hash_u32(unsigned a) {
    a ^= a >> 16; a *= 0x7FEB352Du; a ^= a >> 15; a *= 0x846CA68Bu; a ^= a >> 16;
    return a;
}
// hash-based gumbel noise (JAX threefry is not reproducible here; argmax-structure preserved)
__device__ __forceinline__ float gumbel_noise(unsigned r, unsigned c) {
    unsigned s = hash_u32(r * 0x9E3779B9u ^ c * 0x85EBCA6Bu ^ 1234u);
    float u = ((float)s + 1.0f) * (1.0f / 4294967296.0f);  // (0,1)
    return -__logf(-__logf(u));
}

__device__ __forceinline__ float wave_reduce_sum(float v) {
#pragma unroll
    for (int m = 16; m; m >>= 1) v += __shfl_xor(v, m, 32);
    return v;
}
__device__ __forceinline__ float block_reduce_sum(float v) {
    __shared__ float red[32];
    __syncthreads();                      // allow repeated use
    const int lane = threadIdx.x & 31, w = threadIdx.x >> 5;
    v = wave_reduce_sum(v);
    if (lane == 0) red[w] = v;
    __syncthreads();
    const int nw = blockDim.x >> 5;
    v = (threadIdx.x < (unsigned)nw) ? red[threadIdx.x] : 0.0f;
    if (w == 0) v = wave_reduce_sum(v);
    return v;                             // valid in thread 0
}

// ---------------------------------------------------------------------------
// WMMA fragment loaders (bf16 16x16x32 ISA layout; all chunks 16B/32B aligned)
//  A (16x32, row-major src):
//    lane = 16*hiK + m ; elems 0..7 <- K = 8*hiK+0..7 ; elems 8..15 <- K = 16+8*hiK+0..7
//  B (32x16, src is B^T row-major [N, ld]):
//    lane = 16*hiK + n ; elems 0..7 <- K = 8*hiK+0..7 ; elems 8..15 <- K = 16+8*hiK+0..7
// ---------------------------------------------------------------------------
__device__ __forceinline__ v16bf load_a_frag(const __bf16* rowp, int hiK) {
    v8bf lo = *(const v8bf*)(rowp + hiK * 8);
    v8bf hh = *(const v8bf*)(rowp + 16 + hiK * 8);
    return __builtin_shufflevector(lo, hh, 0,1,2,3,4,5,6,7,8,9,10,11,12,13,14,15);
}
__device__ __forceinline__ v16bf load_b_frag(const __bf16* colp, int hiK) {
    v8bf lo = *(const v8bf*)(colp + hiK * 8);
    v8bf hh = *(const v8bf*)(colp + 16 + hiK * 8);
    return __builtin_shufflevector(lo, hh, 0,1,2,3,4,5,6,7,8,9,10,11,12,13,14,15);
}
// same layouts out of an LDS tile with 32-element rows
__device__ __forceinline__ v16bf lds_a_frag(const __bf16* tile, int row, int hiK) {
    const __bf16* p = tile + row * 32;
    v8bf lo = *(const v8bf*)(p + hiK * 8);
    v8bf hh = *(const v8bf*)(p + 16 + hiK * 8);
    return __builtin_shufflevector(lo, hh, 0,1,2,3,4,5,6,7,8,9,10,11,12,13,14,15);
}
__device__ __forceinline__ v16bf lds_b_frag(const __bf16* tile, int row, int hiK) {
    const __bf16* p = tile + row * 32;
    v8bf lo = *(const v8bf*)(p + hiK * 8);
    v8bf hh = *(const v8bf*)(p + 16 + hiK * 8);
    return __builtin_shufflevector(lo, hh, 0,1,2,3,4,5,6,7,8,9,10,11,12,13,14,15);
}

// ---------------------------------------------------------------------------
// TDM: 2-D tile (tile_d0 x tile_d1, innermost = dim0 = contiguous K) -> LDS
// Descriptor packing per CDNA5 ISA ch.8 (D# group0 / group1; groups 2,3 = 0).
// This toolchain exposes the 6-arg builtin form:
//   (uint32x4 g0, int32x8 g1, int32x4 g2, int32x4 g3, int32x8 g4, i32 cpol)
// ---------------------------------------------------------------------------
__device__ __forceinline__ void tdm_load_2d(unsigned lds_addr, const __bf16* gptr,
                                            unsigned tensor_d0, unsigned tensor_d1,
                                            unsigned tile_d0, unsigned tile_d1,
                                            unsigned stride0_elems) {
    unsigned long long ga = (unsigned long long)(uintptr_t)gptr;
    v4u g0;
    g0[0] = 1u;                                        // count=1, user mode, no gather
    g0[1] = lds_addr;                                  // LDS byte address
    g0[2] = (unsigned)(ga & 0xFFFFFFFFull);            // global_addr[31:0]
    g0[3] = (unsigned)((ga >> 32) & 0x1FFFFFFull) | (2u << 30);  // global_addr[56:32] | type=2
    v8i g1;
    g1[0] = (int)(1u << 16);                           // wg_mask=0, data_size=1 (2 bytes)
    g1[1] = (int)((tensor_d0 & 0xFFFFu) << 16);        // tensor_dim0[15:0] @ bits 63:48
    g1[2] = (int)(((tensor_d0 >> 16) & 0xFFFFu) | ((tensor_d1 & 0xFFFFu) << 16));
    g1[3] = (int)(((tensor_d1 >> 16) & 0xFFFFu) | (tile_d0 << 16));   // tile_dim0 @ 127:112
    g1[4] = (int)(tile_d1 & 0xFFFFu);                  // tile_dim1 @ 143:128 (tile_dim2 = 0)
    g1[5] = (int)stride0_elems;                        // tensor_dim0_stride[31:0] @ 191:160
    g1[6] = 0;                                         // stride0 hi16 = 0, stride1 lo16 = 0
    g1[7] = 0;
    v4i gz4 = {0, 0, 0, 0};                            // 2-D tensor: groups 2/3 unused
    v8i gz8 = {0, 0, 0, 0, 0, 0, 0, 0};
    __builtin_amdgcn_tensor_load_to_lds(g0, g1, gz4, gz4, gz8, 0);
}

// ---------------------------------------------------------------------------
// pack / precompute kernels
// ---------------------------------------------------------------------------
__global__ void init_packs_kernel(unsigned long long* bmu, unsigned long long* zmax) {
    int i = blockIdx.x * blockDim.x + threadIdx.x;
    if (i < B_) { bmu[i] = ~0ull; zmax[i] = 0ull; }
}

__global__ void f32_to_bf16_kernel(const float* __restrict__ in, __bf16* __restrict__ out, int n) {
    int i = blockIdx.x * blockDim.x + threadIdx.x;
    if (i < n) out[i] = to_bf16(in[i]);
}

// W1 [D,H] -> w1t [HP, D] (transposed, padded with zeros)
__global__ void pack_w1t_kernel(const float* __restrict__ W1, __bf16* __restrict__ w1t) {
    int i = blockIdx.x * blockDim.x + threadIdx.x;   // i = n*D + k
    if (i >= HP_ * D_) return;
    int n = i / D_, k = i % D_;
    w1t[i] = (n < H_) ? to_bf16(W1[k * H_ + n]) : to_bf16(0.0f);
}
// W2 [H,K] -> w2t [K, HP] (transposed, padded with zeros)
__global__ void pack_w2t_kernel(const float* __restrict__ W2, __bf16* __restrict__ w2t) {
    int i = blockIdx.x * blockDim.x + threadIdx.x;   // i = n*HP + k
    if (i >= K_ * HP_) return;
    int n = i / HP_, k = i % HP_;
    w2t[i] = (k < H_) ? to_bf16(W2[(size_t)k * K_ + n]) : to_bf16(0.0f);
}

__global__ __launch_bounds__(256) void rowsq_kernel(const float* __restrict__ a,
                                                    float* __restrict__ out, int cols) {
    const int row = blockIdx.x;
    const float* p = a + (size_t)row * cols;
    float s = 0.0f;
    for (int j = threadIdx.x; j < cols; j += 256) { float v = p[j]; s += v * v; }
    s = block_reduce_sum(s);
    if (threadIdx.x == 0) out[row] = s;
}

// ---------------------------------------------------------------------------
// GEMM 1: h = relu(x @ W1 + b1)  -> bf16 [B, HP]
// ---------------------------------------------------------------------------
__global__ __launch_bounds__(384)
void h_gemm_kernel(const __bf16* __restrict__ xb, const __bf16* __restrict__ w1t,
                   const float* __restrict__ b1, __bf16* __restrict__ hb) {
    const int lane = threadIdx.x & 31;
    const int wave = threadIdx.x >> 5;                // 0..11
    const int lm = lane & 15, hi = lane >> 4;
    const int Mb = blockIdx.x * 32 + (wave / 6) * 16;
    const int Nc = (wave % 6) * 16;
    v8f acc = {};
    const __bf16* ap = xb  + (size_t)(Mb + lm) * D_;
    const __bf16* bp = w1t + (size_t)(Nc + lm) * D_;
#pragma unroll 2
    for (int k0 = 0; k0 < D_; k0 += 32) {
        v16bf A  = load_a_frag(ap + k0, hi);
        v16bf Bf = load_b_frag(bp + k0, hi);
        acc = __builtin_amdgcn_wmma_f32_16x16x32_bf16(false, A, false, Bf, (short)0, acc, false, false);
    }
    const int col = Nc + lm;
    const float bias = (col < H_) ? b1[col] : 0.0f;
    union { v8f v; float f[8]; } u; u.v = acc;
#pragma unroll
    for (int e = 0; e < 8; ++e) {
        const int row = Mb + hi * 8 + e;
        float h = u.f[e] + bias;
        hb[(size_t)row * HP_ + col] = to_bf16(h > 0.0f ? h : 0.0f);
    }
}

// ---------------------------------------------------------------------------
// GEMM 2: logits = h @ W2 + b2 (f32 out) + fused argmax((logits+g)/tau)
// block = 8 waves (2x4); wave tile 32x32; block tile 64x128
// ---------------------------------------------------------------------------
__global__ __launch_bounds__(256)
void logits_gemm_kernel(const __bf16* __restrict__ hb, const __bf16* __restrict__ w2t,
                        const float* __restrict__ b2, float* __restrict__ logits,
                        unsigned long long* __restrict__ zmax) {
    const int lane = threadIdx.x & 31;
    const int wave = threadIdx.x >> 5;
    const int lm = lane & 15, hi = lane >> 4;
    const int Mb = blockIdx.y * 64  + (wave >> 2) * 32;
    const int Nb = blockIdx.x * 128 + (wave & 3) * 32;
    v8f acc[2][2] = {};
    const __bf16* a0p = hb  + (size_t)(Mb + lm) * HP_;
    const __bf16* a1p = a0p + (size_t)16 * HP_;
    const __bf16* b0p = w2t + (size_t)(Nb + lm) * HP_;
    const __bf16* b1p = b0p + (size_t)16 * HP_;
#pragma unroll
    for (int k0 = 0; k0 < HP_; k0 += 32) {
        v16bf A0 = load_a_frag(a0p + k0, hi);
        v16bf A1 = load_a_frag(a1p + k0, hi);
        v16bf B0 = load_b_frag(b0p + k0, hi);
        v16bf B1 = load_b_frag(b1p + k0, hi);
        acc[0][0] = __builtin_amdgcn_wmma_f32_16x16x32_bf16(false, A0, false, B0, (short)0, acc[0][0], false, false);
        acc[0][1] = __builtin_amdgcn_wmma_f32_16x16x32_bf16(false, A0, false, B1, (short)0, acc[0][1], false, false);
        acc[1][0] = __builtin_amdgcn_wmma_f32_16x16x32_bf16(false, A1, false, B0, (short)0, acc[1][0], false, false);
        acc[1][1] = __builtin_amdgcn_wmma_f32_16x16x32_bf16(false, A1, false, B1, (short)0, acc[1][1], false, false);
    }
    const unsigned c0 = Nb + lm, c1 = Nb + 16 + lm;
    const float bias0 = b2[c0], bias1 = b2[c1];
#pragma unroll
    for (int mi = 0; mi < 2; ++mi) {
        union { v8f v; float f[8]; } u0, u1;
        u0.v = acc[mi][0]; u1.v = acc[mi][1];
#pragma unroll
        for (int e = 0; e < 8; ++e) {
            const int row = Mb + mi * 16 + hi * 8 + e;
            float l0 = u0.f[e] + bias0;
            float l1 = u1.f[e] + bias1;
            logits[(size_t)row * K_ + c0] = l0;
            logits[(size_t)row * K_ + c1] = l1;
            // tau > 0 never changes the argmax -> divide by tau dropped
            unsigned long long k  = pack_max_key(l0 + gumbel_noise(row, c0), c0);
            unsigned long long k2 = pack_max_key(l1 + gumbel_noise(row, c1), c1);
            if (k2 > k) k = k2;
#pragma unroll
            for (int m = 8; m; m >>= 1) {
                unsigned long long o = __shfl_xor(k, m, 16);
                if (o > k) k = o;
            }
            if (lm == 0) atomicMax(&zmax[row], k);
        }
    }
}

// ---------------------------------------------------------------------------
// GEMM 3 (dominant, compute-bound): dot = x @ codebook^T with TDM-staged LDS
// tiles; fused dist = |x|^2 - 2*dot + |c|^2 and per-row argmin.
// block = 8 waves; block tile 128x128; wave tile 64x32 (8 WMMA / 6 frags).
// Per k-step: TDM loads A(128x32) + B(128x32) bf16 into LDS (16 KB for 1M FLOP).
// ---------------------------------------------------------------------------
__global__ __launch_bounds__(256)
void dist_gemm_kernel(const __bf16* __restrict__ xb, const __bf16* __restrict__ cb,
                      const float* __restrict__ xsq, const float* __restrict__ csq,
                      unsigned long long* __restrict__ bmu) {
    __shared__ __bf16 smA[128 * 32];
    __shared__ __bf16 smB[128 * 32];
    const int lane = threadIdx.x & 31;
    const int wave = threadIdx.x >> 5;
    const int lm = lane & 15, hi = lane >> 4;
    const int wm = wave >> 2;          // 0..1 -> 64-row slab
    const int wn = wave & 3;           // 0..3 -> 32-col slab
    const int Mb = blockIdx.y * 128;
    const int Nb = blockIdx.x * 128;
    v8f acc[4][2] = {};
    const unsigned ldsA = (unsigned)(uintptr_t)(void*)smA;
    const unsigned ldsB = (unsigned)(uintptr_t)(void*)smB;

    for (int k0 = 0; k0 < D_; k0 += 32) {
        __syncthreads();               // previous k-step's LDS consumption done
        if (wave == 0) {
            tdm_load_2d(ldsA, xb + ((size_t)Mb * D_ + k0), D_, B_, 32, 128, D_);
            tdm_load_2d(ldsB, cb + ((size_t)Nb * D_ + k0), D_, K_, 32, 128, D_);
            __builtin_amdgcn_s_wait_tensorcnt(0);
        }
        __syncthreads();               // publish LDS tiles to all waves
        v16bf Af[4], Bf[2];
#pragma unroll
        for (int mi = 0; mi < 4; ++mi) Af[mi] = lds_a_frag(smA, wm * 64 + mi * 16 + lm, hi);
#pragma unroll
        for (int ni = 0; ni < 2; ++ni) Bf[ni] = lds_b_frag(smB, wn * 32 + ni * 16 + lm, hi);
#pragma unroll
        for (int mi = 0; mi < 4; ++mi) {
            acc[mi][0] = __builtin_amdgcn_wmma_f32_16x16x32_bf16(false, Af[mi], false, Bf[0], (short)0, acc[mi][0], false, false);
            acc[mi][1] = __builtin_amdgcn_wmma_f32_16x16x32_bf16(false, Af[mi], false, Bf[1], (short)0, acc[mi][1], false, false);
        }
    }

    const unsigned c0 = Nb + wn * 32 + lm, c1 = c0 + 16;
    const float cs0 = csq[c0], cs1 = csq[c1];
#pragma unroll
    for (int mi = 0; mi < 4; ++mi) {
        union { v8f v; float f[8]; } u0, u1;
        u0.v = acc[mi][0]; u1.v = acc[mi][1];
#pragma unroll
        for (int e = 0; e < 8; ++e) {
            const int row = Mb + wm * 64 + mi * 16 + hi * 8 + e;
            const float xs = xsq[row];
            float d0 = xs - 2.0f * u0.f[e] + cs0;
            float d1 = xs - 2.0f * u1.f[e] + cs1;
            unsigned long long k  = pack_min_key(d0, c0);
            unsigned long long k2 = pack_min_key(d1, c1);
            if (k2 < k) k = k2;
#pragma unroll
            for (int m = 8; m; m >>= 1) {
                unsigned long long o = __shfl_xor(k, m, 16);
                if (o < k) k = o;
            }
            if (lm == 0) atomicMin(&bmu[row], k);
        }
    }
}

// ---------------------------------------------------------------------------
// delta: per-row sum |codebook[z_idx[row]] - x[row]| (f32 gather)
// ---------------------------------------------------------------------------
__global__ __launch_bounds__(256)
void delta_rows_kernel(const float* __restrict__ x, const float* __restrict__ codebook,
                       const unsigned long long* __restrict__ zmax, const int* __restrict__ t,
                       float* __restrict__ rowz, float* __restrict__ rowu) {
    const int row = blockIdx.x;
    const unsigned zc = (0xFFFFFFFFu - (unsigned)(zmax[row] & 0xFFFFFFFFull)) & (K_ - 1);
    const float* xr = x + (size_t)row * D_;
    const float* cr = codebook + (size_t)zc * D_;
    float s = 0.0f;
    for (int j = threadIdx.x; j < D_; j += 256) s += fabsf(cr[j] - xr[j]);
    s = block_reduce_sum(s);
    if (threadIdx.x == 0) rowz[row] = s;

    float su = 0.0f;
    if (*t == 0) {
        const unsigned uidx = hash_u32((unsigned)row * 0x9E3779B9u ^ 5678u) & (K_ - 1);
        const float* ur = codebook + (size_t)uidx * D_;
        for (int j = threadIdx.x; j < D_; j += 256) su += fabsf(ur[j] - xr[j]);
    }
    su = block_reduce_sum(su);
    if (threadIdx.x == 0) rowu[row] = su;
}

__global__ void write_bmu_kernel(const unsigned long long* __restrict__ bmu, float* __restrict__ out) {
    int i = blockIdx.x * blockDim.x + threadIdx.x;
    if (i < B_) out[i] = (float)(unsigned)(bmu[i] & 0xFFFFFFFFull);
}

__global__ __launch_bounds__(256)
void finalize_delta_kernel(const float* __restrict__ rowz, const float* __restrict__ rowu,
                           const int* __restrict__ t, float* __restrict__ out) {
    float s = 0.0f, su = 0.0f;
    for (int i = threadIdx.x; i < B_; i += 256) { s += rowz[i]; su += rowu[i]; }
    s  = block_reduce_sum(s);
    su = block_reduce_sum(su);
    if (threadIdx.x == 0) {
        const float inv = 1.0f / ((float)B_ * (float)D_);
        float mz = s * inv;
        out[0] = (*t == 0) ? (mz * 0.1f + su * inv) : mz;
    }
}

// ---------------------------------------------------------------------------
// host launcher
// ---------------------------------------------------------------------------
extern "C" void kernel_launch(void* const* d_in, const int* in_sizes, int n_in,
                              void* d_out, int out_size, void* d_ws, size_t ws_size,
                              hipStream_t stream) {
    const float* x   = (const float*)d_in[0];
    const float* cbf = (const float*)d_in[1];
    const float* W1  = (const float*)d_in[2];
    const float* b1  = (const float*)d_in[3];
    const float* W2  = (const float*)d_in[4];
    const float* b2  = (const float*)d_in[5];
    const int*   t   = (const int*)d_in[6];
    float* out = (float*)d_out;

    char* ws = (char*)d_ws;
    size_t off = 0;
    auto alloc = [&](size_t bytes) -> void* {
        void* p = ws + off;
        off = (off + bytes + 255) & ~(size_t)255;
        return p;
    };
    __bf16* xb   = (__bf16*)alloc((size_t)B_ * D_ * 2);   // 32 MB
    __bf16* cb   = (__bf16*)alloc((size_t)K_ * D_ * 2);   // 16 MB
    __bf16* w1t  = (__bf16*)alloc((size_t)HP_ * D_ * 2);
    __bf16* w2t  = (__bf16*)alloc((size_t)K_ * HP_ * 2);
    __bf16* hb   = (__bf16*)alloc((size_t)B_ * HP_ * 2);
    float*  xsq  = (float*)alloc((size_t)B_ * 4);
    float*  csq  = (float*)alloc((size_t)K_ * 4);
    unsigned long long* bmu  = (unsigned long long*)alloc((size_t)B_ * 8);
    unsigned long long* zmax = (unsigned long long*)alloc((size_t)B_ * 8);
    float*  rowz = (float*)alloc((size_t)B_ * 4);
    float*  rowu = (float*)alloc((size_t)B_ * 4);

    init_packs_kernel<<<(B_ + 255) / 256, 256, 0, stream>>>(bmu, zmax);
    f32_to_bf16_kernel<<<(B_ * D_ + 255) / 256, 256, 0, stream>>>(x, xb, B_ * D_);
    f32_to_bf16_kernel<<<(K_ * D_ + 255) / 256, 256, 0, stream>>>(cbf, cb, K_ * D_);
    pack_w1t_kernel<<<(HP_ * D_ + 255) / 256, 256, 0, stream>>>(W1, w1t);
    pack_w2t_kernel<<<(K_ * HP_ + 255) / 256, 256, 0, stream>>>(W2, w2t);
    rowsq_kernel<<<B_, 256, 0, stream>>>(x, xsq, D_);
    rowsq_kernel<<<K_, 256, 0, stream>>>(cbf, csq, D_);

    h_gemm_kernel<<<B_ / 32, 384, 0, stream>>>(xb, w1t, b1, hb);
    logits_gemm_kernel<<<dim3(K_ / 128, B_ / 64), 256, 0, stream>>>(hb, w2t, b2, out, zmax);
    dist_gemm_kernel<<<dim3(K_ / 128, B_ / 128), 256, 0, stream>>>(xb, cb, xsq, csq, bmu);

    delta_rows_kernel<<<B_, 256, 0, stream>>>(x, cbf, zmax, t, rowz, rowu);
    write_bmu_kernel<<<(B_ + 255) / 256, 256, 0, stream>>>(bmu, out + (size_t)B_ * K_);
    finalize_delta_kernel<<<1, 256, 0, stream>>>(rowz, rowu, t, out + (size_t)B_ * K_ + B_);
}
